// GATv2StateEncoder_61555471286659
// MI455X (gfx1250) — matrix operand order repaired and checked by reading, not compiled, not run on hardware
//
#include <hip/hip_runtime.h>

// ---------------------------------------------------------------------------
// GATv2 state encoder for MI455X (gfx1250, wave32, WMMA + async-to-LDS).
// Dense node GEMMs: v_wmma_f32_16x16x32_bf16, weights staged into LDS once per
// workgroup via global_load_async_to_lds_b128 (ASYNCcnt), A streamed b128 from
// L2. Edge softmax/aggregation passes are float4-vectorized and L2-resident
// (XL/XR tables ~25.6MB each << 192MB L2), scatter via f32 atomics.
// ---------------------------------------------------------------------------

typedef __attribute__((ext_vector_type(16))) __bf16 v16bf;
typedef __attribute__((ext_vector_type(8)))  float  v8f;

#define NEG_SLOPE 0.2f
#define GK        128          // inner dim of every WMMA GEMM in this net
#define ROWPAD    (GK + 8)     // LDS row stride (bf16 elems): 68-dword lane stride
#define WAVES_PER_BLOCK 4

// ---------------- utility kernels ----------------

__global__ void f32_to_bf16_kernel(const float* __restrict__ src,
                                   __bf16* __restrict__ dst, long n) {
    long i = (long)blockIdx.x * blockDim.x + threadIdx.x;
    if (i < n) dst[i] = (__bf16)src[i];
}

__global__ void fill_f32_kernel(float* __restrict__ p, float v, long n) {
    long i = (long)blockIdx.x * blockDim.x + threadIdx.x;
    if (i < n) p[i] = v;
}

__device__ __forceinline__ void atomicMaxF(float* addr, float val) {
    int* ia = (int*)addr;
    int cur = __float_as_int(*addr);
    while (__int_as_float(cur) < val) {
        int prev = atomicCAS(ia, cur, __float_as_int(val));
        if (prev == cur) break;
        cur = prev;
    }
}

// ---------------- WMMA GEMM: C[M,Nout] = A[M,GK] * W[Nout,GK]^T + bias --------
// Block = 4 waves. Waves share one LDS-staged weight slice of 16*NT output
// rows (async copied, ASYNCcnt). Each wave owns one 16-row tile of A and
// computes NT 16x16 output tiles from the shared slice.

template <int NT>
__global__ void wmma_gemm_smem(const __bf16* __restrict__ A,
                               const __bf16* __restrict__ W,
                               const float* __restrict__ bias,
                               float* __restrict__ Cf,     // optional f32 output
                               __bf16* __restrict__ Cb,    // optional bf16 output
                               int M, int Nout, int doRelu) {
    extern __shared__ char smem_raw[];
    __bf16* smem = (__bf16*)smem_raw;                    // [16*NT][ROWPAD]

    const int tid    = threadIdx.x;                      // 0..127
    const int lane   = tid & 31;
    const int wave   = tid >> 5;
    const int l15    = lane & 15;
    const int half   = lane >> 4;
    const int n0base = blockIdx.y * (16 * NT);

    // ---- stage weight slice into LDS with async global->LDS copies ----
    // 16*NT rows of GK bf16, 16B chunks; padded rows avoid ds bank conflicts.
    for (int cidx = tid; cidx < NT * 16 * (GK / 8); cidx += WAVES_PER_BLOCK * 32) {
        int row = cidx >> 4;                             // GK/8 == 16 chunks per row
        int ck  = cidx & 15;
        unsigned ldsaddr = (unsigned)(uintptr_t)(smem + row * ROWPAD + ck * 8);
        const __bf16* g  = W + (size_t)(n0base + row) * GK + ck * 8;
        asm volatile("global_load_async_to_lds_b128 %0, %1, off"
                     :: "v"(ldsaddr), "v"(g) : "memory");
    }
    asm volatile("s_wait_asynccnt 0" ::: "memory");      // this wave's copies landed
    __syncthreads();                                     // all waves' copies landed

    // ---- accumulate ----
    const int rowTile = blockIdx.x * WAVES_PER_BLOCK + wave;
    const int m0 = rowTile * 16;
    const int ar = min(m0 + l15, M - 1);                 // clamp; masked on store
    const __bf16* arow = A + (size_t)ar * GK;

    v8f acc[NT];
#pragma unroll
    for (int t = 0; t < NT; ++t) acc[t] = (v8f){};

#pragma unroll
    for (int k0 = 0; k0 < GK; k0 += 32) {
        v16bf a;
        // A 16x32 bf16 fragment: lane holds row l15, K[half*16..+15]
        ((uint4*)&a)[0] = *(const uint4*)(arow + k0 + half * 16);
        ((uint4*)&a)[1] = *(const uint4*)(arow + k0 + half * 16 + 8);
#pragma unroll
        for (int t = 0; t < NT; ++t) {
            // B 32x16 fragment from LDS: lane holds col (t*16+l15), K[half*16..+15]
            const __bf16* brow = smem + (t * 16 + l15) * ROWPAD + k0 + half * 16;
            v16bf b;
            ((uint4*)&b)[0] = *(const uint4*)(brow);
            ((uint4*)&b)[1] = *(const uint4*)(brow + 8);
            acc[t] = __builtin_amdgcn_wmma_f32_16x16x32_bf16(
                false, a, false, b, (short)0, acc[t], false, false);
        }
    }

    // ---- epilogue: C layout VGPR r -> M = r + 8*half, N = lane&15 ----
#pragma unroll
    for (int t = 0; t < NT; ++t) {
        const int col = n0base + t * 16 + l15;
        const float bv = bias ? bias[col] : 0.0f;
#pragma unroll
        for (int r = 0; r < 8; ++r) {
            int row = m0 + r + half * 8;
            if (row >= M) continue;
            float v = acc[t][r] + bv;
            if (doRelu) v = v > 0.0f ? v : 0.0f;
            size_t o = (size_t)row * Nout + col;
            if (Cf) Cf[o] = v;
            if (Cb) Cb[o] = (__bf16)v;
        }
    }
}

// ---------------- edge pass A: logits + per-dst max ----------------

__global__ void edge_logits_max(const float* __restrict__ XL, const float* __restrict__ XR,
                                const float* __restrict__ att,
                                const int* __restrict__ esrc, const int* __restrict__ edst,
                                int E, int Nn, int heads, int C,
                                float* __restrict__ elog, float* __restrict__ nmax) {
    int idx = blockIdx.x * blockDim.x + threadIdx.x;
    int total = (E + Nn) * heads;
    if (idx >= total) return;
    int e = idx / heads;
    int h = idx - e * heads;
    int s = (e < E) ? esrc[e] : (e - E);   // appended self loops
    int d = (e < E) ? edst[e] : (e - E);
    const float4* xl = (const float4*)(XL + ((size_t)s * heads + h) * C);
    const float4* xr = (const float4*)(XR + ((size_t)d * heads + h) * C);
    const float4* at = (const float4*)(att + (size_t)h * C);
    float acc = 0.0f;
    for (int c = 0; c < (C >> 2); ++c) {
        float4 a = xl[c], b = xr[c], w = at[c];
        float v;
        v = a.x + b.x; v = v > 0.0f ? v : NEG_SLOPE * v; acc += v * w.x;
        v = a.y + b.y; v = v > 0.0f ? v : NEG_SLOPE * v; acc += v * w.y;
        v = a.z + b.z; v = v > 0.0f ? v : NEG_SLOPE * v; acc += v * w.z;
        v = a.w + b.w; v = v > 0.0f ? v : NEG_SLOPE * v; acc += v * w.w;
    }
    elog[idx] = acc;
    atomicMaxF(&nmax[(size_t)d * heads + h], acc);
}

// ---------------- edge pass B: exp(logit - max) + per-dst sum ----------------

__global__ void edge_exp_sum(const int* __restrict__ edst, int E, int Nn, int heads,
                             const float* __restrict__ nmax,
                             float* __restrict__ elog, float* __restrict__ nsum) {
    int idx = blockIdx.x * blockDim.x + threadIdx.x;
    int total = (E + Nn) * heads;
    if (idx >= total) return;
    int e = idx / heads;
    int h = idx - e * heads;
    int d = (e < E) ? edst[e] : (e - E);
    float ex = __expf(elog[idx] - nmax[(size_t)d * heads + h]);
    elog[idx] = ex;
    atomicAdd(&nsum[(size_t)d * heads + h], ex);
}

// ---------------- edge pass C: out[dst] += alpha * XL[src], 4 floats/thread ----

__global__ void edge_scatter4(const float* __restrict__ XL, const float* __restrict__ elog,
                              const float* __restrict__ nsum,
                              const int* __restrict__ esrc, const int* __restrict__ edst,
                              int E, int Nn, int heads, int C,
                              float* __restrict__ acc) {
    long idx = (long)blockIdx.x * blockDim.x + threadIdx.x;
    const int F = heads * C;
    const int Q = F >> 2;                 // float4 groups per edge
    long total = (long)(E + Nn) * Q;
    if (idx >= total) return;
    long e = idx / Q;
    int q = (int)(idx - e * Q);
    int f0 = q * 4;
    int h = f0 / C;                       // C multiple of 4 -> same head for all 4
    int s = (e < E) ? esrc[e] : (int)(e - E);
    int d = (e < E) ? edst[e] : (int)(e - E);
    float alpha = elog[e * heads + h] / nsum[(size_t)d * heads + h];
    float4 xv = *(const float4*)(XL + (size_t)s * F + f0);
    float* dp = acc + (size_t)d * F + f0;
    atomicAdd(dp + 0, alpha * xv.x);
    atomicAdd(dp + 1, alpha * xv.y);
    atomicAdd(dp + 2, alpha * xv.z);
    atomicAdd(dp + 3, alpha * xv.w);
}

// ---------------- bias + ReLU, output as bf16 (feeds next WMMA GEMM) ----------

__global__ void bias_relu_bf16(const float* __restrict__ acc, const float* __restrict__ bias,
                               int F, long total, __bf16* __restrict__ outb) {
    long i = (long)blockIdx.x * blockDim.x + threadIdx.x;
    if (i >= total) return;
    int c = (int)(i % F);
    float v = acc[i] + bias[c];
    v = v > 0.0f ? v : 0.0f;
    outb[i] = (__bf16)v;
}

// ---------------- bias + ReLU + mean-pool partial reduce (C==32, block 256) ----

__global__ void bias_relu_pool32(const float* __restrict__ acc, const float* __restrict__ bias,
                                 int Nn, float* __restrict__ pooled) {
    __shared__ float part[256];
    const int c = threadIdx.x & 31;
    const int g = threadIdx.x >> 5;      // 8 node-rows per block iteration
    float local = 0.0f;
    for (int n = blockIdx.x * 8 + g; n < Nn; n += gridDim.x * 8) {
        float v = acc[(size_t)n * 32 + c] + bias[c];
        local += v > 0.0f ? v : 0.0f;
    }
    part[threadIdx.x] = local;
    __syncthreads();
    if (threadIdx.x < 32) {
        float s = part[threadIdx.x];
#pragma unroll
        for (int gg = 1; gg < 8; ++gg) s += part[gg * 32 + threadIdx.x];
        atomicAdd(&pooled[threadIdx.x], s);
    }
}

// ---------------- final 32 -> 96 GEMV on pooled mean ----------------

__global__ void final_linear(const float* __restrict__ pooled, const float* __restrict__ Wout,
                             const float* __restrict__ bout, float invN,
                             float* __restrict__ out) {
    int o = blockIdx.x * blockDim.x + threadIdx.x;
    if (o >= 96) return;
    float s = bout[o];
#pragma unroll
    for (int c = 0; c < 32; ++c) s += Wout[o * 32 + c] * (pooled[c] * invN);
    out[o] = s;
}

// ---------------------------------------------------------------------------

extern "C" void kernel_launch(void* const* d_in, const int* in_sizes, int n_in,
                              void* d_out, int out_size, void* d_ws, size_t ws_size,
                              hipStream_t stream) {
    const float* x     = (const float*)d_in[0];
    const int*   eidx  = (const int*)d_in[1];
    // d_in[2] = batch (all zeros, single graph) -> mean pool over all nodes
    const float* Win   = (const float*)d_in[3];
    const float* b_in  = (const float*)d_in[4];
    const float* Wl1   = (const float*)d_in[5];
    const float* bl1   = (const float*)d_in[6];
    const float* Wr1   = (const float*)d_in[7];
    const float* br1   = (const float*)d_in[8];
    const float* att1  = (const float*)d_in[9];
    const float* bias1 = (const float*)d_in[10];
    const float* Wl2   = (const float*)d_in[11];
    const float* bl2   = (const float*)d_in[12];
    const float* Wr2   = (const float*)d_in[13];
    const float* br2   = (const float*)d_in[14];
    const float* att2  = (const float*)d_in[15];
    const float* bias2 = (const float*)d_in[16];
    const float* Wout  = (const float*)d_in[17];
    const float* bout  = (const float*)d_in[18];
    float* out = (float*)d_out;

    const int Nn = in_sizes[0] / 128;   // 50000
    const int E  = in_sizes[1] / 2;     // 600000
    const int ET = E + Nn;              // + self loops
    const int* esrc = eidx;
    const int* edst = eidx + E;

    // -------- workspace carve-out (256B aligned) --------
    char* base = (char*)d_ws;
    size_t off = 0;
    auto take = [&](size_t bytes) -> char* {
        char* p = base + off;
        off = (off + bytes + 255) & ~(size_t)255;
        return p;
    };
    __bf16* Xbf   = (__bf16*)take((size_t)Nn * 128 * 2);
    __bf16* WinB  = (__bf16*)take(128 * 128 * 2);
    __bf16* Wl1B  = (__bf16*)take(128 * 128 * 2);
    __bf16* Wr1B  = (__bf16*)take(128 * 128 * 2);
    __bf16* Wl2B  = (__bf16*)take(32 * 128 * 2);
    __bf16* Wr2B  = (__bf16*)take(32 * 128 * 2);
    __bf16* Hbf   = (__bf16*)take((size_t)Nn * 128 * 2);
    float*  XL1   = (float*)take((size_t)Nn * 128 * 4);
    float*  XR1   = (float*)take((size_t)Nn * 128 * 4);
    float*  elog1 = (float*)take((size_t)ET * 2 * 4);
    float*  nmax1 = (float*)take((size_t)Nn * 2 * 4);
    float*  nsum1 = (float*)take((size_t)Nn * 2 * 4);
    float*  XL2   = (float*)take((size_t)Nn * 32 * 4);
    float*  XR2   = (float*)take((size_t)Nn * 32 * 4);
    float*  elog2 = (float*)take((size_t)ET * 4);
    float*  nmax2 = (float*)take((size_t)Nn * 4);
    float*  nsum2 = (float*)take((size_t)Nn * 4);
    float*  pooled= (float*)take(32 * 4);
    // Disjoint-lifetime aliases (zeroed before reuse, all ordering on `stream`):
    float*  acc1  = XR1;   // XR1 dead after edge pass A of conv1
    float*  acc2  = XR2;   // XR2 dead after edge pass A of conv2
    __bf16* H1bf  = Xbf;   // Xbf dead after first GEMM

    const int TB = 256;
    const long nx = (long)Nn * 128;

    // -------- precision conversion (f32 -> bf16) --------
    f32_to_bf16_kernel<<<(nx + TB - 1) / TB, TB, 0, stream>>>(x, Xbf, nx);
    f32_to_bf16_kernel<<<(16384 + TB - 1) / TB, TB, 0, stream>>>(Win, WinB, 16384);
    f32_to_bf16_kernel<<<(16384 + TB - 1) / TB, TB, 0, stream>>>(Wl1, Wl1B, 16384);
    f32_to_bf16_kernel<<<(16384 + TB - 1) / TB, TB, 0, stream>>>(Wr1, Wr1B, 16384);
    f32_to_bf16_kernel<<<(4096 + TB - 1) / TB, TB, 0, stream>>>(Wl2, Wl2B, 4096);
    f32_to_bf16_kernel<<<(4096 + TB - 1) / TB, TB, 0, stream>>>(Wr2, Wr2B, 4096);

    // -------- per-launch init (deterministic; capture-safe) --------
    fill_f32_kernel<<<(Nn * 2 + TB - 1) / TB, TB, 0, stream>>>(nmax1, -3.0e38f, (long)Nn * 2);
    fill_f32_kernel<<<(Nn + TB - 1) / TB, TB, 0, stream>>>(nmax2, -3.0e38f, (long)Nn);
    hipMemsetAsync(nsum1, 0, (size_t)Nn * 2 * 4, stream);
    hipMemsetAsync(nsum2, 0, (size_t)Nn * 4, stream);
    hipMemsetAsync(pooled, 0, 32 * 4, stream);

    // -------- lin_in + conv1 projections (WMMA bf16, LDS-staged weights) -----
    const int rowTiles  = (Nn + 15) / 16;
    const int rowBlocks = (rowTiles + WAVES_PER_BLOCK - 1) / WAVES_PER_BLOCK;
    const size_t smem4 = (size_t)(4 * 16) * ROWPAD * 2;   // NT=4 slice
    const size_t smem2 = (size_t)(2 * 16) * ROWPAD * 2;   // NT=2 slice
    dim3 g128(rowBlocks, 128 / (16 * 4));                  // Nout=128, NT=4 -> y=2
    wmma_gemm_smem<4><<<g128, 128, smem4, stream>>>(Xbf, WinB, b_in, nullptr, Hbf, Nn, 128, 0);
    wmma_gemm_smem<4><<<g128, 128, smem4, stream>>>(Hbf, Wl1B, bl1, XL1, nullptr, Nn, 128, 0);
    wmma_gemm_smem<4><<<g128, 128, smem4, stream>>>(Hbf, Wr1B, br1, XR1, nullptr, Nn, 128, 0);

    // -------- conv1 edge softmax + aggregate (heads=2, C=64) --------
    const int tot1 = ET * 2;
    edge_logits_max<<<(tot1 + TB - 1) / TB, TB, 0, stream>>>(
        XL1, XR1, att1, esrc, edst, E, Nn, 2, 64, elog1, nmax1);
    hipMemsetAsync(acc1, 0, (size_t)Nn * 128 * 4, stream);   // reuse of XR1 begins here
    edge_exp_sum<<<(tot1 + TB - 1) / TB, TB, 0, stream>>>(edst, E, Nn, 2, nmax1, elog1, nsum1);
    const long sc1 = (long)ET * (128 / 4);
    edge_scatter4<<<(sc1 + TB - 1) / TB, TB, 0, stream>>>(
        XL1, elog1, nsum1, esrc, edst, E, Nn, 2, 64, acc1);
    bias_relu_bf16<<<(nx + TB - 1) / TB, TB, 0, stream>>>(acc1, bias1, 128, nx, H1bf);

    // -------- conv2 projections (WMMA bf16, Nout=32, NT=2) --------
    dim3 g32(rowBlocks, 32 / (16 * 2));                    // y=1
    wmma_gemm_smem<2><<<g32, 128, smem2, stream>>>(H1bf, Wl2B, bl2, XL2, nullptr, Nn, 32, 0);
    wmma_gemm_smem<2><<<g32, 128, smem2, stream>>>(H1bf, Wr2B, br2, XR2, nullptr, Nn, 32, 0);

    // -------- conv2 edge softmax + aggregate (heads=1, C=32) --------
    const int tot2 = ET;
    edge_logits_max<<<(tot2 + TB - 1) / TB, TB, 0, stream>>>(
        XL2, XR2, att2, esrc, edst, E, Nn, 1, 32, elog2, nmax2);
    hipMemsetAsync(acc2, 0, (size_t)Nn * 32 * 4, stream);    // reuse of XR2 begins here
    edge_exp_sum<<<(tot2 + TB - 1) / TB, TB, 0, stream>>>(edst, E, Nn, 1, nmax2, elog2, nsum2);
    const long sc2 = (long)ET * (32 / 4);
    edge_scatter4<<<(sc2 + TB - 1) / TB, TB, 0, stream>>>(
        XL2, elog2, nsum2, esrc, edst, E, Nn, 1, 32, acc2);

    // -------- bias+ReLU + mean pool + final linear --------
    bias_relu_pool32<<<512, 256, 0, stream>>>(acc2, bias2, Nn, pooled);
    final_linear<<<1, 128, 0, stream>>>(pooled, Wout, bout, 1.0f / (float)Nn, out);
}
